// SelfAttention_90074054131982
// MI455X (gfx1250) — matrix operand order repaired and checked by reading, not compiled, not run on hardware
//
#include <hip/hip_runtime.h>

// ---------------------------------------------------------------------------
// Self-attention (B=2, S=2048, D=2048, H=16, dh=128) for gfx1250 / MI455X.
// Compute-bound (206 GFLOP vs ~0.25 GB HBM traffic) -> bf16 WMMA
// (v_wmma_f32_16x16x32_bf16, f32 accumulate).
// Operand fragments as loaded here (wave32):
//   A (16x32, 16-bit): lane m=L%16; elems 0-15 at k = c*32 + (L/16)*16
//                      (one contiguous 32B run, same k order as B -> 2x b128)
//   B (32x16, 16-bit): lane n=L%16; elems 0-15 at k = c*32 + (L/16)*16
//                      (one contiguous 32B run -> 2x b128)
//   C/D (16x16 f32):   lane n=L%16; vgpr r -> m = r + 8*(L/16)
// GEMMs are software-pipelined (next-chunk fragments prefetched before this
// chunk's WMMAs) to avoid s_wait_loadcnt 0 in front of every v_wmma.
// ---------------------------------------------------------------------------

typedef __attribute__((ext_vector_type(16))) __bf16        v16bf;
typedef __attribute__((ext_vector_type(8)))  float         v8f;
typedef __attribute__((ext_vector_type(4)))  unsigned int  v4u;

union BFrag {
    v16bf bf;
    v4u   u[2];   // 32 bytes = 16 bf16
};

__device__ __forceinline__ v8f wmma_bf16(const BFrag& a, const BFrag& b, v8f c) {
    return __builtin_amdgcn_wmma_f32_16x16x32_bf16(
        false, a.bf, false, b.bf, (short)0, c, false, false);
}

__device__ __forceinline__ void load_afrag(BFrag& f, const unsigned short* p) {
    // p already includes + (lane/16)*16 ; 16 contiguous bf16
    f.u[0] = *(const v4u*)(p);
    f.u[1] = *(const v4u*)(p + 8);
}
__device__ __forceinline__ void load_bfrag(BFrag& f, const unsigned short* p) {
    // p already includes + (lane/16)*16 ; 16 contiguous bf16
    f.u[0] = *(const v4u*)(p);
    f.u[1] = *(const v4u*)(p + 8);
}

__device__ __forceinline__ unsigned short f2bf(float x) {
    unsigned int u = __float_as_uint(x);
    unsigned int r = u + 0x7FFFu + ((u >> 16) & 1u);   // RNE
    return (unsigned short)(r >> 16);
}

// ------------------------------ f32 -> bf16 --------------------------------
__global__ __launch_bounds__(256) void cvt_bf16_kernel(
    const float* __restrict__ in, unsigned short* __restrict__ out, int n)
{
    int i = blockIdx.x * blockDim.x + threadIdx.x;
    int stride = gridDim.x * blockDim.x;
    for (; i < n; i += stride) out[i] = f2bf(in[i]);
}

// --------------------- QKV projection (+RoPE) in bf16 ----------------------
// C[m,n] = sum_k A[m,k] * W[n,k];  M=4096, N=2048, K=2048.
// Wave tile: 32(M) x 64(N); block = 8 waves -> 256(M) x 64(N).
__global__ __launch_bounds__(256) void qkv_proj_kernel(
    const unsigned short* __restrict__ A,      // [4096, 2048] bf16 (x)
    const unsigned short* __restrict__ W,      // [2048, 2048] bf16
    const float*          __restrict__ fcos,   // [2048, 64]
    const float*          __restrict__ fsin,   // [2048, 64]
    unsigned short*       __restrict__ Out,
    int rope, int transposeOut)
{
    const int lane = threadIdx.x & 31;
    const int w    = threadIdx.x >> 5;
    const int ln   = lane & 15;
    const int lh   = lane >> 4;
    const int mbase = blockIdx.y * 256 + w * 32;
    const int nbase = blockIdx.x * 64;

    v8f acc[2][4] = {};
    const unsigned short* arow0 = A + (size_t)(mbase + ln) * 2048 + lh * 16;
    const unsigned short* arow1 = arow0 + 16 * 2048;
    const unsigned short* brow[4];
#pragma unroll
    for (int nt = 0; nt < 4; ++nt)
        brow[nt] = W + (size_t)(nbase + nt * 16 + ln) * 2048 + lh * 16;

    BFrag af0, af1, bf[4];
    load_afrag(af0, arow0);
    load_afrag(af1, arow1);
#pragma unroll
    for (int nt = 0; nt < 4; ++nt) load_bfrag(bf[nt], brow[nt]);

    for (int kc = 0; kc < 2048; kc += 32) {
        const int kn = (kc + 32) & 2047;       // wraps on last iter (harmless)
        BFrag af0n, af1n, bfn[4];
        load_afrag(af0n, arow0 + kn);
        load_afrag(af1n, arow1 + kn);
#pragma unroll
        for (int nt = 0; nt < 4; ++nt) load_bfrag(bfn[nt], brow[nt] + kn);
#pragma unroll
        for (int nt = 0; nt < 4; ++nt) acc[0][nt] = wmma_bf16(af0, bf[nt], acc[0][nt]);
#pragma unroll
        for (int nt = 0; nt < 4; ++nt) acc[1][nt] = wmma_bf16(af1, bf[nt], acc[1][nt]);
        af0 = af0n; af1 = af1n;
#pragma unroll
        for (int nt = 0; nt < 4; ++nt) bf[nt] = bfn[nt];
    }

    const int H = 16, DH = 128, S = 2048;
#pragma unroll
    for (int mt = 0; mt < 2; ++mt) {
#pragma unroll
        for (int nt = 0; nt < 4; ++nt) {
#pragma unroll
            for (int r = 0; r < 8; ++r) {
                int   n   = nbase + nt * 16 + ln;
                int   row = mbase + mt * 16 + r + 8 * lh;
                int   b   = row >> 11;
                int   s   = row & 2047;
                float v   = acc[mt][nt][r];
                if (rope) {
                    float partner = __shfl_xor(v, 1, 32);
                    int   i  = (n & 127) >> 1;
                    float c  = fcos[s * 64 + i];
                    float sn = fsin[s * 64 + i];
                    v = (n & 1) ? (partner * sn + v * c)
                                : (v * c - partner * sn);
                }
                int h = n >> 7, dh = n & 127;
                size_t addr = transposeOut
                    ? ((size_t)(b * H + h) * DH + dh) * S + s     // VT [B,H,dh,S]
                    : ((size_t)(b * H + h) * S + s) * DH + dh;    // Q/K [B,H,S,dh]
                Out[addr] = f2bf(v);
            }
        }
    }
}

// ----------------------- flash attention per head --------------------------
// grid = B*H*(S/128); 8 waves, each wave: 16 queries x dh=128.
__global__ __launch_bounds__(256) void attn_kernel(
    const unsigned short* __restrict__ Q,    // [B,H,S,dh]
    const unsigned short* __restrict__ K,    // [B,H,S,dh]
    const unsigned short* __restrict__ VT,   // [B,H,dh,S]
    unsigned short*       __restrict__ O)    // [B,S,H*dh]
{
    __shared__ unsigned short pbuf[8][16 * 32];   // per-wave P staging (1 KB)

    const int lane = threadIdx.x & 31;
    const int w    = threadIdx.x >> 5;
    const int ln   = lane & 15;
    const int lh   = lane >> 4;

    const int S = 2048, DH = 128, H = 16;
    const int bid = blockIdx.x;
    const int qc = bid & 15, h = (bid >> 4) & 15, b = bid >> 8;
    const int qbase = qc * 128 + w * 16;

    const size_t head = (size_t)(b * H + h) * S * DH;

    // Q: 16 queries x 128 dh = 4 resident A-fragments
    BFrag qf[4];
    const unsigned short* qrow = Q + head + (size_t)(qbase + ln) * DH + lh * 16;
#pragma unroll
    for (int c = 0; c < 4; ++c) load_afrag(qf[c], qrow + c * 32);

    float mrun[8], lrun[8];
    v8f   acc[8] = {};
#pragma unroll
    for (int r = 0; r < 8; ++r) { mrun[r] = -1e30f; lrun[r] = 0.0f; }

    const float scale = 0.08838834764831845f;   // 1/sqrt(128)
    unsigned short* prow = &pbuf[w][0];

    for (int kb = 0; kb < S; kb += 32) {
        // ---- scores: keys kb..kb+15 (s0), kb+16..kb+31 (s1)
        v8f s0 = {}, s1 = {};
        const unsigned short* krow0 = K + head + (size_t)(kb + ln) * DH + lh * 16;
        const unsigned short* krow1 = krow0 + 16 * DH;
#pragma unroll
        for (int c = 0; c < 4; ++c) {
            BFrag k0, k1;
            load_bfrag(k0, krow0 + c * 32);
            load_bfrag(k1, krow1 + c * 32);
            s0 = wmma_bf16(qf[c], k0, s0);
            s1 = wmma_bf16(qf[c], k1, s1);
        }

        // ---- issue V loads now; softmax VALU work below hides their latency
        BFrag vf[8];
#pragma unroll
        for (int t = 0; t < 8; ++t) {
            const unsigned short* vrow =
                VT + head + (size_t)(t * 16 + ln) * S + kb + lh * 16;
            load_bfrag(vf[t], vrow);
        }

        // ---- online softmax; row m = r + 8*lh spans the 16 lanes of a half
        float alpha[8];
#pragma unroll
        for (int r = 0; r < 8; ++r) {
            float a0 = s0[r] * scale, a1 = s1[r] * scale;
            float mx = fmaxf(a0, a1);
#pragma unroll
            for (int off = 8; off >= 1; off >>= 1)
                mx = fmaxf(mx, __shfl_xor(mx, off, 32));
            float mnew = fmaxf(mrun[r], mx);
            alpha[r]   = __expf(mrun[r] - mnew);
            mrun[r]    = mnew;
            float p0 = __expf(a0 - mnew), p1 = __expf(a1 - mnew);
            float rs = p0 + p1;
#pragma unroll
            for (int off = 8; off >= 1; off >>= 1)
                rs += __shfl_xor(rs, off, 32);
            lrun[r] = lrun[r] * alpha[r] + rs;
            int m = r + 8 * lh;
            prow[m * 32 + ln]      = f2bf(p0);
            prow[m * 32 + 16 + ln] = f2bf(p1);
        }
#pragma unroll
        for (int t = 0; t < 8; ++t)
#pragma unroll
            for (int r = 0; r < 8; ++r) acc[t][r] *= alpha[r];

        __builtin_amdgcn_wave_barrier();   // DS stores before DS loads

        // P as a 16x32 A operand (two ds_load_b128 per lane)
        BFrag pf;
        pf.u[0] = *(const v4u*)(prow + ln * 32 + lh * 16);
        pf.u[1] = *(const v4u*)(prow + ln * 32 + lh * 16 + 8);

        // ---- acc += P(16x32) * V(32 keys x 128 dh)
#pragma unroll
        for (int t = 0; t < 8; ++t) acc[t] = wmma_bf16(pf, vf[t], acc[t]);
    }

    // ---- epilogue: normalize, store O[b, s, h*128+dh] (bf16)
#pragma unroll
    for (int t = 0; t < 8; ++t) {
#pragma unroll
        for (int r = 0; r < 8; ++r) {
            float v   = acc[t][r] / lrun[r];
            int   row = qbase + r + 8 * lh;
            int   col = h * DH + t * 16 + ln;
            O[((size_t)b * S + row) * 2048 + col] = f2bf(v);
        }
    }
}

// --------------------------- output projection -----------------------------
// Same pipelined 32x64 wave tile; f32 output to d_out.
__global__ __launch_bounds__(256) void out_proj_kernel(
    const unsigned short* __restrict__ A,   // [4096, 2048] bf16 (attn out)
    const unsigned short* __restrict__ W,   // [2048, 2048] bf16 (wo)
    float* __restrict__ Out)                // [4096, 2048] f32
{
    const int lane = threadIdx.x & 31;
    const int w    = threadIdx.x >> 5;
    const int ln   = lane & 15;
    const int lh   = lane >> 4;
    const int mbase = blockIdx.y * 256 + w * 32;
    const int nbase = blockIdx.x * 64;

    v8f acc[2][4] = {};
    const unsigned short* arow0 = A + (size_t)(mbase + ln) * 2048 + lh * 16;
    const unsigned short* arow1 = arow0 + 16 * 2048;
    const unsigned short* brow[4];
#pragma unroll
    for (int nt = 0; nt < 4; ++nt)
        brow[nt] = W + (size_t)(nbase + nt * 16 + ln) * 2048 + lh * 16;

    BFrag af0, af1, bf[4];
    load_afrag(af0, arow0);
    load_afrag(af1, arow1);
#pragma unroll
    for (int nt = 0; nt < 4; ++nt) load_bfrag(bf[nt], brow[nt]);

    for (int kc = 0; kc < 2048; kc += 32) {
        const int kn = (kc + 32) & 2047;
        BFrag af0n, af1n, bfn[4];
        load_afrag(af0n, arow0 + kn);
        load_afrag(af1n, arow1 + kn);
#pragma unroll
        for (int nt = 0; nt < 4; ++nt) load_bfrag(bfn[nt], brow[nt] + kn);
#pragma unroll
        for (int nt = 0; nt < 4; ++nt) acc[0][nt] = wmma_bf16(af0, bf[nt], acc[0][nt]);
#pragma unroll
        for (int nt = 0; nt < 4; ++nt) acc[1][nt] = wmma_bf16(af1, bf[nt], acc[1][nt]);
        af0 = af0n; af1 = af1n;
#pragma unroll
        for (int nt = 0; nt < 4; ++nt) bf[nt] = bfn[nt];
    }
#pragma unroll
    for (int mt = 0; mt < 2; ++mt)
#pragma unroll
        for (int nt = 0; nt < 4; ++nt)
#pragma unroll
            for (int r = 0; r < 8; ++r) {
                int n   = nbase + nt * 16 + ln;
                int row = mbase + mt * 16 + r + 8 * lh;
                Out[(size_t)row * 2048 + n] = acc[mt][nt][r];
            }
}

// ------------------------------- launcher ----------------------------------
extern "C" void kernel_launch(void* const* d_in, const int* in_sizes, int n_in,
                              void* d_out, int out_size, void* d_ws, size_t ws_size,
                              hipStream_t stream)
{
    (void)in_sizes; (void)n_in; (void)out_size; (void)ws_size;

    const float* x  = (const float*)d_in[0];
    const float* wq = (const float*)d_in[1];
    const float* wk = (const float*)d_in[2];
    const float* wv = (const float*)d_in[3];
    const float* wo = (const float*)d_in[4];
    const float* fc = (const float*)d_in[5];
    const float* fs = (const float*)d_in[6];

    unsigned short* ws  = (unsigned short*)d_ws;
    const size_t NX = (size_t)4096 * 2048;
    const size_t NW = (size_t)2048 * 2048;
    unsigned short* Xb  = ws;
    unsigned short* Wqb = Xb  + NX;
    unsigned short* Wkb = Wqb + NW;
    unsigned short* Wvb = Wkb + NW;
    unsigned short* Wob = Wvb + NW;
    unsigned short* Qb  = Wob + NW;          // [B,H,S,dh]
    unsigned short* Kb  = Qb  + NX;          // [B,H,S,dh]
    unsigned short* VTb = Kb  + NX;          // [B,H,dh,S]
    unsigned short* Ob  = VTb + NX;          // [B,S,D]

    cvt_bf16_kernel<<<1024, 256, 0, stream>>>(x,  Xb,  (int)NX);
    cvt_bf16_kernel<<<1024, 256, 0, stream>>>(wq, Wqb, (int)NW);
    cvt_bf16_kernel<<<1024, 256, 0, stream>>>(wk, Wkb, (int)NW);
    cvt_bf16_kernel<<<1024, 256, 0, stream>>>(wv, Wvb, (int)NW);
    cvt_bf16_kernel<<<1024, 256, 0, stream>>>(wo, Wob, (int)NW);

    dim3 pgrid(2048 / 64, 4096 / 256);   // (32, 16)
    qkv_proj_kernel<<<pgrid, 256, 0, stream>>>(Xb, Wqb, fc, fs, Qb,  1, 0);
    qkv_proj_kernel<<<pgrid, 256, 0, stream>>>(Xb, Wkb, fc, fs, Kb,  1, 0);
    qkv_proj_kernel<<<pgrid, 256, 0, stream>>>(Xb, Wvb, fc, fs, VTb, 0, 1);

    attn_kernel<<<512, 256, 0, stream>>>(Qb, Kb, VTb, Ob);

    out_proj_kernel<<<pgrid, 256, 0, stream>>>(Ob, Wob, (float*)d_out);
}
